// Block_65352222376027
// MI455X (gfx1250) — compile-verified, matrix operand order checked by reading, not run on hardware
//
#include <hip/hip_runtime.h>

// ---------------------------------------------------------------------------
// RWKV6 block for MI455X (gfx1250): bf16 WMMA GEMMs (double-buffered LDS,
// float4 global loads, global_prefetch) + segment-parallel WKV scan.
// ---------------------------------------------------------------------------

#define T_DIM 2048
#define C_DIM 2048
#define H_DIM 32
#define HS_DIM 64
#define FFN_DIM 7168
#define NS_DIM 4

typedef __attribute__((ext_vector_type(16))) __bf16 v16bf;
typedef __attribute__((ext_vector_type(8)))  float  v8f;

// Native fp32->bf16 conversion (gfx1250 has hardware bf16 converts, RNE).
__device__ __forceinline__ __bf16 f2bf(float f) { return (__bf16)f; }

// ---------------------------------------------------------------------------
// GEMM: C[M,N] = epilogue(A[M,K] @ B[K,N])  (row-major, fp32 in/out, bf16 MMA)
// EPI: 0=none 1=silu 2=relu^2 3=sigmoid 4=add base 5=tanh
// Block tile 128x128, K-tile 64, double-buffered LDS. 256 threads = 8 waves
// (4 x 2); each wave owns a 32x64 region = 2x4 WMMA tiles of 16x16.
// ---------------------------------------------------------------------------
template <int EPI, bool GUARD>
__global__ __launch_bounds__(256)
void gemm_bf16_kernel(const float* __restrict__ A, int lda,
                      const float* __restrict__ B, int ldb,
                      float* __restrict__ C, int ldc,
                      const float* __restrict__ base,
                      int M, int N, int K) {
    __shared__ __bf16 As[2][128][72];   // [buf][m][k]
    __shared__ __bf16 Bs[2][128][72];   // [buf][n][k]  (transposed)

    const int tid    = threadIdx.x;
    const int lane   = tid & 31;
    const int waveId = tid >> 5;
    const int waveM  = waveId >> 1;      // 0..3
    const int waveN  = waveId & 1;       // 0..1
    const int colIn  = lane & 15;
    const int half   = lane >> 4;

    const int m0 = blockIdx.y * 128;
    const int n0 = blockIdx.x * 128;

    v8f acc[2][4] = {};
    float4 aS[8], bS[8];

    // issue float4 global loads of one 128x64 A tile + 64x128 B tile into regs
    auto load_tile_regs = [&](int k0) {
        #pragma unroll
        for (int t = 0; t < 8; t++) {
            int i  = tid + t * 256;               // 0..2047
            // A: 128 rows x 16 float4
            int rA = i >> 4, cA = (i & 15) << 2;
            int gm = m0 + rA, gk = k0 + cA;
            if (!GUARD) {
                aS[t] = *(const float4*)(A + (size_t)gm * lda + gk);
            } else {
                float4 v = make_float4(0.f, 0.f, 0.f, 0.f);
                if (gm < M) {
                    if (gk + 3 < K) v = *(const float4*)(A + (size_t)gm * lda + gk);
                    else {
                        float e0 = (gk + 0 < K) ? A[(size_t)gm * lda + gk + 0] : 0.f;
                        float e1 = (gk + 1 < K) ? A[(size_t)gm * lda + gk + 1] : 0.f;
                        float e2 = (gk + 2 < K) ? A[(size_t)gm * lda + gk + 2] : 0.f;
                        float e3 = (gk + 3 < K) ? A[(size_t)gm * lda + gk + 3] : 0.f;
                        v = make_float4(e0, e1, e2, e3);
                    }
                }
                aS[t] = v;
            }
            // B: 64 rows x 32 float4
            int rB = i >> 5, cB = (i & 31) << 2;
            int gkb = k0 + rB, gn = n0 + cB;
            if (!GUARD) {
                bS[t] = *(const float4*)(B + (size_t)gkb * ldb + gn);
            } else {
                float4 v = make_float4(0.f, 0.f, 0.f, 0.f);
                if (gkb < K) {
                    if (gn + 3 < N) v = *(const float4*)(B + (size_t)gkb * ldb + gn);
                    else {
                        float e0 = (gn + 0 < N) ? B[(size_t)gkb * ldb + gn + 0] : 0.f;
                        float e1 = (gn + 1 < N) ? B[(size_t)gkb * ldb + gn + 1] : 0.f;
                        float e2 = (gn + 2 < N) ? B[(size_t)gkb * ldb + gn + 2] : 0.f;
                        float e3 = (gn + 3 < N) ? B[(size_t)gkb * ldb + gn + 3] : 0.f;
                        v = make_float4(e0, e1, e2, e3);
                    }
                }
                bS[t] = v;
            }
        }
    };

    auto store_tile_lds = [&](int buf) {
        #pragma unroll
        for (int t = 0; t < 8; t++) {
            int i  = tid + t * 256;
            int rA = i >> 4, cA = (i & 15) << 2;
            __bf16* pa = &As[buf][rA][cA];
            pa[0] = f2bf(aS[t].x); pa[1] = f2bf(aS[t].y);
            pa[2] = f2bf(aS[t].z); pa[3] = f2bf(aS[t].w);
            int rB = i >> 5, cB = (i & 31) << 2;
            Bs[buf][cB + 0][rB] = f2bf(bS[t].x);
            Bs[buf][cB + 1][rB] = f2bf(bS[t].y);
            Bs[buf][cB + 2][rB] = f2bf(bS[t].z);
            Bs[buf][cB + 3][rB] = f2bf(bS[t].w);
        }
    };

    const int nIter = (K + 63) >> 6;

    load_tile_regs(0);
    store_tile_lds(0);
    __syncthreads();

    for (int it = 0; it < nIter; ++it) {
        const int cur = it & 1;

        // prefetch K-tile two steps ahead into the cache hierarchy
        if (it + 2 < nIter) {
            int k2 = (it + 2) << 6;
            __builtin_prefetch(A + (size_t)(m0 + (tid >> 1)) * lda + k2, 0, 1);
            __builtin_prefetch(B + (size_t)(k2 + (tid >> 2)) * ldb + n0, 0, 1);
        }
        // start next tile's global loads (overlap with WMMA below)
        if (it + 1 < nIter) load_tile_regs((it + 1) << 6);

        #pragma unroll
        for (int ks = 0; ks < 64; ks += 32) {
            v16bf af[2], bfr[4];
            #pragma unroll
            for (int tm = 0; tm < 2; tm++) {
                int m = waveM * 32 + tm * 16 + colIn;
                const __bf16* p0 = &As[cur][m][ks + half * 8];
                const __bf16* p1 = &As[cur][m][ks + 16 + half * 8];
                #pragma unroll
                for (int e = 0; e < 8; e++) { af[tm][e] = p0[e]; af[tm][8 + e] = p1[e]; }
            }
            #pragma unroll
            for (int tn = 0; tn < 4; tn++) {
                int n = waveN * 64 + tn * 16 + colIn;
                const __bf16* p0 = &Bs[cur][n][ks + half * 8];
                const __bf16* p1 = &Bs[cur][n][ks + 16 + half * 8];
                #pragma unroll
                for (int e = 0; e < 8; e++) { bfr[tn][e] = p0[e]; bfr[tn][8 + e] = p1[e]; }
            }
            #pragma unroll
            for (int tm = 0; tm < 2; tm++)
                #pragma unroll
                for (int tn = 0; tn < 4; tn++)
                    acc[tm][tn] = __builtin_amdgcn_wmma_f32_16x16x32_bf16(
                        false, af[tm], false, bfr[tn],
                        (short)0, acc[tm][tn], false, false);
        }

        if (it + 1 < nIter) store_tile_lds(cur ^ 1);
        __syncthreads();
    }

    // --- epilogue + store ---
    #pragma unroll
    for (int tm = 0; tm < 2; tm++) {
        #pragma unroll
        for (int tn = 0; tn < 4; tn++) {
            int gm0 = m0 + waveM * 32 + tm * 16 + half * 8;
            int gn  = n0 + waveN * 64 + tn * 16 + colIn;
            if (GUARD && gn >= N) continue;
            #pragma unroll
            for (int r = 0; r < 8; r++) {
                int gm = gm0 + r;
                if (GUARD && gm >= M) continue;
                float v = acc[tm][tn][r];
                if (EPI == 1) v = v / (1.0f + __expf(-v));                 // silu
                else if (EPI == 2) { v = v > 0.0f ? v : 0.0f; v = v * v; } // relu^2
                else if (EPI == 3) v = 1.0f / (1.0f + __expf(-v));         // sigmoid
                else if (EPI == 4) v += base[(size_t)gm * ldc + gn];       // residual
                else if (EPI == 5) v = tanhf(v);
                C[(size_t)gm * ldc + gn] = v;
            }
        }
    }
}

// ---------------------------------------------------------------------------
// LayerNorm over last dim (C=2048), one block per token.
// ---------------------------------------------------------------------------
__global__ __launch_bounds__(256)
void ln_kernel(const float* __restrict__ x, const float* __restrict__ g,
               const float* __restrict__ b, float* __restrict__ out, float eps) {
    int t = blockIdx.x;
    const float* xr = x + (size_t)t * C_DIM;
    float* orow = out + (size_t)t * C_DIM;
    float s = 0.f, s2 = 0.f;
    for (int c = threadIdx.x; c < C_DIM; c += 256) { float v = xr[c]; s += v; s2 += v * v; }
    __shared__ float rs[256], rq[256];
    rs[threadIdx.x] = s; rq[threadIdx.x] = s2;
    __syncthreads();
    for (int off = 128; off > 0; off >>= 1) {
        if (threadIdx.x < off) { rs[threadIdx.x] += rs[threadIdx.x + off]; rq[threadIdx.x] += rq[threadIdx.x + off]; }
        __syncthreads();
    }
    float mean = rs[0] * (1.0f / C_DIM);
    float var  = rq[0] * (1.0f / C_DIM) - mean * mean;
    float inv  = rsqrtf(var + eps);
    for (int c = threadIdx.x; c < C_DIM; c += 256)
        orow[c] = (xr[c] - mean) * inv * g[c] + b[c];
}

// xx = shift(xa, start_pos, state) - xa
__global__ void shift_diff_kernel(const float* __restrict__ xa,
                                  const int* __restrict__ start_pos,
                                  const float* __restrict__ state,
                                  float* __restrict__ xx, int total) {
    int idx = blockIdx.x * 256 + threadIdx.x;
    if (idx >= total) return;
    int t = idx >> 11, c = idx & (C_DIM - 1);
    float sh = (t > 0) ? xa[idx - C_DIM] : 0.0f;
    #pragma unroll
    for (int s = 0; s < NS_DIM; s++)
        if (t == start_pos[s]) sh = state[s * C_DIM + c];
    xx[idx] = sh - xa[idx];
}

// out = xa + xx * (maa[c] + mix[idx])   (mix may be null)
__global__ void mixadd_kernel(const float* __restrict__ xa, const float* __restrict__ xx,
                              const float* __restrict__ maa, const float* __restrict__ mix,
                              float* __restrict__ out, int total) {
    int idx = blockIdx.x * 256 + threadIdx.x;
    if (idx >= total) return;
    int c = idx & (C_DIM - 1);
    float m = maa[c] + (mix ? mix[idx] : 0.0f);
    out[idx] = xa[idx] + xx[idx] * m;
}

// D = exp(-exp(td[c] + D))
__global__ void dec_kernel(const float* __restrict__ td, float* __restrict__ D, int total) {
    int idx = blockIdx.x * 256 + threadIdx.x;
    if (idx >= total) return;
    int c = idx & (C_DIM - 1);
    D[idx] = __expf(-__expf(td[c] + D[idx]));
}

// ---------------------------------------------------------------------------
// WKV scan. The recurrence overwrites S at each sequence start, so the NS
// packed segments are independent: parallelize over (segment, head) = 128
// blocks of 64 threads; thread j owns state column j in 64 VGPRs.
// ---------------------------------------------------------------------------
__global__ __launch_bounds__(64)
void wkv_scan_kernel(const float* __restrict__ r, const float* __restrict__ k,
                     const float* __restrict__ v, const float* __restrict__ dec,
                     const float* __restrict__ u, const float* __restrict__ wkv_state,
                     const int* __restrict__ start_pos, float* __restrict__ y) {
    int h = blockIdx.x & (H_DIM - 1);
    int s = blockIdx.x >> 5;                 // 0..NS-1
    int j = threadIdx.x;                     // 0..63
    int sp      = start_pos[s];
    int t_begin = (s == 0) ? 0 : sp;
    int t_end   = (s + 1 < NS_DIM) ? start_pos[s + 1] : T_DIM;
    if (t_end > T_DIM) t_end = T_DIM;

    __shared__ __align__(16) float sr[64], sk[64], sv[64], sd[64], su[64];
    float S[64];
    #pragma unroll
    for (int i = 0; i < 64; i++) S[i] = 0.0f;
    su[j] = u[h * 64 + j];
    __syncthreads();

    for (int t = t_begin; t < t_end; t++) {
        int base = (t * H_DIM + h) * HS_DIM;
        sr[j] = r[base + j]; sk[j] = k[base + j];
        sv[j] = v[base + j]; sd[j] = dec[base + j];
        if (t == sp) {
            const float* Wst = wkv_state + ((size_t)(s * H_DIM + h) * HS_DIM) * HS_DIM;
            #pragma unroll 8
            for (int i = 0; i < 64; i++) S[i] = Wst[i * 64 + j];
        }
        __syncthreads();
        float vj = sv[j];
        float yj = 0.0f, bonus = 0.0f;
        #pragma unroll
        for (int i4 = 0; i4 < 16; i4++) {
            float4 r4 = *(const float4*)&sr[i4 * 4];
            float4 u4 = *(const float4*)&su[i4 * 4];
            float4 k4 = *(const float4*)&sk[i4 * 4];
            float4 d4 = *(const float4*)&sd[i4 * 4];
            int i = i4 * 4;
            yj += r4.x * S[i + 0] + r4.y * S[i + 1] + r4.z * S[i + 2] + r4.w * S[i + 3];
            bonus += r4.x * u4.x * k4.x + r4.y * u4.y * k4.y
                   + r4.z * u4.z * k4.z + r4.w * u4.w * k4.w;
            S[i + 0] = d4.x * S[i + 0] + k4.x * vj;
            S[i + 1] = d4.y * S[i + 1] + k4.y * vj;
            S[i + 2] = d4.z * S[i + 2] + k4.z * vj;
            S[i + 3] = d4.w * S[i + 3] + k4.w * vj;
        }
        y[base + j] = yj + bonus * vj;
        __syncthreads();
    }
}

// GroupNorm(H groups of 64) * g  -> a (input to Wo GEMM)
__global__ __launch_bounds__(64)
void gnorm_mul_kernel(const float* __restrict__ y, const float* __restrict__ g,
                      const float* __restrict__ gn_g, const float* __restrict__ gn_b,
                      float* __restrict__ out) {
    int th = blockIdx.x;
    int j = threadIdx.x;
    int idx = th * 64 + j;
    int c = idx & (C_DIM - 1);
    float v = y[idx];
    __shared__ float rs[64], rq[64];
    rs[j] = v; rq[j] = v * v;
    __syncthreads();
    for (int off = 32; off > 0; off >>= 1) {
        if (j < off) { rs[j] += rs[j + off]; rq[j] += rq[j + off]; }
        __syncthreads();
    }
    float mean = rs[0] * (1.0f / 64.0f);
    float var  = rq[0] * (1.0f / 64.0f) - mean * mean;
    float inv  = rsqrtf(var + 6.4e-4f);                // 1e-5 * 8^2
    out[idx] = ((v - mean) * inv * gn_g[c] + gn_b[c]) * g[idx];
}

// out = xmid + sig * kv
__global__ void final_kernel(const float* __restrict__ xmid, const float* __restrict__ sig,
                             const float* __restrict__ kv, float* __restrict__ out, int total) {
    int idx = blockIdx.x * 256 + threadIdx.x;
    if (idx >= total) return;
    out[idx] = xmid[idx] + sig[idx] * kv[idx];
}

// ---------------------------------------------------------------------------
template <int EPI>
static void gemm_launch(const float* A, int lda, const float* B, int ldb,
                        float* C, int ldc, const float* base,
                        int M, int N, int K, hipStream_t stream) {
    dim3 grid((N + 127) / 128, (M + 127) / 128), blk(256);
    bool full = (M % 128 == 0) && (N % 128 == 0) && (K % 64 == 0);
    if (full)
        gemm_bf16_kernel<EPI, false><<<grid, blk, 0, stream>>>(A, lda, B, ldb, C, ldc, base, M, N, K);
    else
        gemm_bf16_kernel<EPI, true><<<grid, blk, 0, stream>>>(A, lda, B, ldb, C, ldc, base, M, N, K);
}

extern "C" void kernel_launch(void* const* d_in, const int* in_sizes, int n_in,
                              void* d_out, int out_size, void* d_ws, size_t ws_size,
                              hipStream_t stream) {
    const float* x         = (const float*)d_in[0];
    const int*   start_pos = (const int*)  d_in[1];
    const float* att_state = (const float*)d_in[2];
    const float* wkv_state = (const float*)d_in[3];
    const float* ffn_state = (const float*)d_in[4];
    const float* ln1_g = (const float*)d_in[5],  *ln1_b = (const float*)d_in[6];
    const float* ln2_g = (const float*)d_in[7],  *ln2_b = (const float*)d_in[8];
    const float* tm_maa_x = (const float*)d_in[9];
    const float* tm_maa_w = (const float*)d_in[10];
    const float* tm_maa_k = (const float*)d_in[11];
    const float* tm_maa_v = (const float*)d_in[12];
    const float* tm_maa_r = (const float*)d_in[13];
    const float* tm_maa_g = (const float*)d_in[14];
    const float* tm_w1  = (const float*)d_in[15];   // [C,160]
    const float* tm_w2  = (const float*)d_in[16];   // [5,32,C]
    const float* td     = (const float*)d_in[17];
    const float* td_w1  = (const float*)d_in[18];   // [C,64]
    const float* td_w2  = (const float*)d_in[19];   // [64,C]
    const float* u      = (const float*)d_in[20];
    const float* Wr = (const float*)d_in[21], *Wk = (const float*)d_in[22];
    const float* Wv = (const float*)d_in[23], *Wg = (const float*)d_in[24];
    const float* Wo = (const float*)d_in[25];
    const float* lnx_g = (const float*)d_in[26], *lnx_b = (const float*)d_in[27];
    const float* cm_maa_k = (const float*)d_in[28], *cm_maa_r = (const float*)d_in[29];
    const float* Wck = (const float*)d_in[30];      // [C,FFN]
    const float* Wcv = (const float*)d_in[31];      // [FFN,C]
    const float* Wcr = (const float*)d_in[32];

    float* W = (float*)d_ws;
    const size_t TC = (size_t)T_DIM * C_DIM;
    float* XA = W + 0 * TC;          // ln output / xf
    float* XX = W + 1 * TC;          // shift diff
    float* Ab = W + 2 * TC;          // mixed activation / y / xk2
    float* Mb = W + 3 * TC;          // mix_s / a / xr2
    float* Db = W + 4 * TC;          // dec
    float* Kb = W + 5 * TC;          // k / kv
    float* Vb = W + 6 * TC;          // v
    float* Rb = W + 7 * TC;          // r / sigmoid(xr2@Wcr)
    float* Gb = W + 8 * TC;          // silu(xg@Wg)
    float* X2 = W + 9 * TC;          // x after attention residual
    float* Hb = W + 10 * TC;         // relu^2(xk2@Wck) [T,FFN]
    float* T160 = Hb + (size_t)T_DIM * FFN_DIM;
    float* T64  = T160 + (size_t)T_DIM * 160;

    const int TOT = T_DIM * C_DIM;
    const int ewg = (TOT + 255) / 256;
    dim3 blk(256);

    // ---------------- Tmix ----------------
    ln_kernel<<<T_DIM, 256, 0, stream>>>(x, ln1_g, ln1_b, XA, 1e-5f);
    shift_diff_kernel<<<ewg, blk, 0, stream>>>(XA, start_pos, att_state, XX, TOT);
    mixadd_kernel<<<ewg, blk, 0, stream>>>(XA, XX, tm_maa_x, nullptr, Ab, TOT);
    gemm_launch<5>(Ab, C_DIM, tm_w1, 160, T160, 160, nullptr,
                   T_DIM, 160, C_DIM, stream);                       // tanh(z@tm_w1)

    // s = 0 : w path (decay)
    gemm_launch<0>(T160 + 0 * 32, 160, tm_w2 + 0 * 32 * C_DIM, C_DIM,
                   Mb, C_DIM, nullptr, T_DIM, C_DIM, 32, stream);
    mixadd_kernel<<<ewg, blk, 0, stream>>>(XA, XX, tm_maa_w, Mb, Ab, TOT);       // xw
    gemm_launch<5>(Ab, C_DIM, td_w1, 64, T64, 64, nullptr,
                   T_DIM, 64, C_DIM, stream);                        // tanh(xw@td_w1)
    gemm_launch<0>(T64, 64, td_w2, C_DIM, Db, C_DIM, nullptr,
                   T_DIM, C_DIM, 64, stream);
    dec_kernel<<<ewg, blk, 0, stream>>>(td, Db, TOT);                // exp(-exp(td + .))

    // s = 1 : k
    gemm_launch<0>(T160 + 1 * 32, 160, tm_w2 + 1 * 32 * C_DIM, C_DIM,
                   Mb, C_DIM, nullptr, T_DIM, C_DIM, 32, stream);
    mixadd_kernel<<<ewg, blk, 0, stream>>>(XA, XX, tm_maa_k, Mb, Ab, TOT);
    gemm_launch<0>(Ab, C_DIM, Wk, C_DIM, Kb, C_DIM, nullptr,
                   T_DIM, C_DIM, C_DIM, stream);

    // s = 2 : v
    gemm_launch<0>(T160 + 2 * 32, 160, tm_w2 + 2 * 32 * C_DIM, C_DIM,
                   Mb, C_DIM, nullptr, T_DIM, C_DIM, 32, stream);
    mixadd_kernel<<<ewg, blk, 0, stream>>>(XA, XX, tm_maa_v, Mb, Ab, TOT);
    gemm_launch<0>(Ab, C_DIM, Wv, C_DIM, Vb, C_DIM, nullptr,
                   T_DIM, C_DIM, C_DIM, stream);

    // s = 3 : r
    gemm_launch<0>(T160 + 3 * 32, 160, tm_w2 + 3 * 32 * C_DIM, C_DIM,
                   Mb, C_DIM, nullptr, T_DIM, C_DIM, 32, stream);
    mixadd_kernel<<<ewg, blk, 0, stream>>>(XA, XX, tm_maa_r, Mb, Ab, TOT);
    gemm_launch<0>(Ab, C_DIM, Wr, C_DIM, Rb, C_DIM, nullptr,
                   T_DIM, C_DIM, C_DIM, stream);

    // s = 4 : g (silu epilogue)
    gemm_launch<0>(T160 + 4 * 32, 160, tm_w2 + 4 * 32 * C_DIM, C_DIM,
                   Mb, C_DIM, nullptr, T_DIM, C_DIM, 32, stream);
    mixadd_kernel<<<ewg, blk, 0, stream>>>(XA, XX, tm_maa_g, Mb, Ab, TOT);
    gemm_launch<1>(Ab, C_DIM, Wg, C_DIM, Gb, C_DIM, nullptr,
                   T_DIM, C_DIM, C_DIM, stream);

    // WKV recurrence, parallel over (segment, head); y -> Ab
    wkv_scan_kernel<<<NS_DIM * H_DIM, 64, 0, stream>>>(Rb, Kb, Vb, Db, u,
                                                       wkv_state, start_pos, Ab);

    // GroupNorm * g -> Mb, then residual GEMM: X2 = x + Mb @ Wo
    gnorm_mul_kernel<<<T_DIM * H_DIM, 64, 0, stream>>>(Ab, Gb, lnx_g, lnx_b, Mb);
    gemm_launch<4>(Mb, C_DIM, Wo, C_DIM, X2, C_DIM, x,
                   T_DIM, C_DIM, C_DIM, stream);

    // ---------------- CMix ----------------
    ln_kernel<<<T_DIM, 256, 0, stream>>>(X2, ln2_g, ln2_b, XA, 1e-5f);
    shift_diff_kernel<<<ewg, blk, 0, stream>>>(XA, start_pos, ffn_state, XX, TOT);
    mixadd_kernel<<<ewg, blk, 0, stream>>>(XA, XX, cm_maa_k, nullptr, Ab, TOT);  // xk2
    mixadd_kernel<<<ewg, blk, 0, stream>>>(XA, XX, cm_maa_r, nullptr, Mb, TOT);  // xr2
    gemm_launch<2>(Ab, C_DIM, Wck, FFN_DIM, Hb, FFN_DIM, nullptr,
                   T_DIM, FFN_DIM, C_DIM, stream);                   // relu^2
    gemm_launch<0>(Hb, FFN_DIM, Wcv, C_DIM, Kb, C_DIM, nullptr,
                   T_DIM, C_DIM, FFN_DIM, stream);                   // kv
    gemm_launch<3>(Mb, C_DIM, Wcr, C_DIM, Rb, C_DIM, nullptr,
                   T_DIM, C_DIM, C_DIM, stream);                     // sigmoid
    final_kernel<<<ewg, blk, 0, stream>>>(X2, Rb, Kb, (float*)d_out, TOT);
}
